// LSTMSensorFusion_57647051047606
// MI455X (gfx1250) — compile-verified, matrix operand order checked by reading, not run on hardware
//
#include <hip/hip_runtime.h>
#include <hip/hip_bf16.h>

// ---------------------------------------------------------------------------
// LSTM sensor fusion, MI455X (gfx1250).
//   B=512, T=2048, N_SENS=32, H=64, L=4, OUT=3.
// Strategy:
//   K1: transpose Wg -> bf16 [L][256][128]   (WMMA B-matrix friendly)
//   K2: fused input projection per token -> bf16 x [T][B][H]
//   K3: persistent LSTM: 32 workgroups x (16 batch rows), all weights in LDS
//       (256KB bf16), gate GEMMs via v_wmma_f32_16x16x32_bf16, x_t staged via
//       global_load_async_to_lds, load-all-then-compute K loop, 8 waves/WG.
//   K4: fused output projection per token -> fp32 y [B][T][3]
// ---------------------------------------------------------------------------

#define BATCH   512
#define TSTEPS  2048
#define NSENS   32
#define HDIM    64
#define NLAY    4
#define ODIM    3

typedef unsigned short ushort_t;
typedef __attribute__((ext_vector_type(16))) __bf16 v16bf;
typedef __attribute__((ext_vector_type(8)))  float  v8f;

struct U8q { uint4 lo, hi; };            // 32B container for one A/B tile slice
struct H4  { unsigned short a[4]; };     // 4 packed bf16 -> uint2

// Workspace layout (bytes)
#define XIN_OFF   ((size_t)0)                                   // bf16 [T][B][H]
#define XIN_BYTES ((size_t)TSTEPS * BATCH * HDIM * 2)           // 128 MB
#define H3_OFF    (XIN_OFF + XIN_BYTES)                         // bf16 [B][T][H]
#define H3_BYTES  ((size_t)BATCH * TSTEPS * HDIM * 2)           // 128 MB
#define WT_OFF    (H3_OFF + H3_BYTES)                           // bf16 [L][256][128]
#define WT_BYTES  ((size_t)NLAY * 256 * 128 * 2)

// Dynamic LDS partition for the LSTM kernel
#define LDS_W      (NLAY * 256 * 128 * 2)          // 262144 weights bf16
#define LDS_AL     (NLAY * 16 * 128 * 2)           //  16384 [inp|h] bf16 per layer
#define LDS_C      (NLAY * 16 * 64 * 4)            //  16384 cell state f32
#define LDS_G      (16 * 256 * 4)                  //  16384 gates f32
#define LDS_BG     (NLAY * 256 * 4)                //   4096 biases f32
#define SMEM_BYTES (LDS_W + LDS_AL + LDS_C + LDS_G + LDS_BG)   // 315392 (308 KB)

__device__ __forceinline__ unsigned short f2bf(float f) {
    unsigned int u = __float_as_uint(f);
    u += 0x7FFFu + ((u >> 16) & 1u);     // round to nearest even
    return (unsigned short)(u >> 16);
}
__device__ __forceinline__ float bf2f(unsigned short h) {
    return __uint_as_float(((unsigned int)h) << 16);
}
__device__ __forceinline__ float sigmoid_f(float x) {
    return 1.0f / (1.0f + __expf(-x));
}
__device__ __forceinline__ float tanh_fast(float x) {
    x = fminf(fmaxf(x, -15.0f), 15.0f);
    const float e = __expf(2.0f * x);
    return (e - 1.0f) / (e + 1.0f);
}
__device__ __forceinline__ float gelu_f(float x) {
    return 0.5f * x * (1.0f + erff(x * 0.70710678118654752f));
}
// Load one 16x32 bf16 tile slice for this lane (A and B share the layout):
// halves 0..7 <- k = base+0..7, halves 8..15 <- k = base+16..23
__device__ __forceinline__ v16bf ldAB(const ushort_t* p) {
    U8q t;
    t.lo = *(const uint4*)(p);
    t.hi = *(const uint4*)(p + 16);
    return __builtin_bit_cast(v16bf, t);
}

// ---------------------------------------------------------------------------
// K1: Wg fp32 [L][128][256] -> bf16 transposed [L][256][128]
// ---------------------------------------------------------------------------
__global__ void prep_weights_kernel(const float* __restrict__ Wg,
                                    ushort_t* __restrict__ wt) {
    int idx = blockIdx.x * 256 + threadIdx.x;   // 131072 total
    int k = idx & 127;
    int n = (idx >> 7) & 255;
    int l = idx >> 15;
    wt[idx] = f2bf(Wg[((size_t)(l * 128 + k)) * 256 + n]);
}

// ---------------------------------------------------------------------------
// K2: x = GELU(LN(sensors @ W1 + b1)) @ W2 + b2  -> bf16 [T][B][H]
// ---------------------------------------------------------------------------
__global__ void __launch_bounds__(128) input_proj_kernel(
        const float* __restrict__ sensors,
        const float* __restrict__ W1, const float* __restrict__ b1,
        const float* __restrict__ g1, const float* __restrict__ bt1,
        const float* __restrict__ W2, const float* __restrict__ b2,
        ushort_t* __restrict__ xin) {
    __shared__ float sW1[NSENS * HDIM];
    __shared__ float sW2[HDIM * HDIM];
    __shared__ float sb1[HDIM], sg[HDIM], sb[HDIM], sb2[HDIM];
    const int tid = threadIdx.x;
    for (int i = tid; i < NSENS * HDIM; i += 128) sW1[i] = W1[i];
    for (int i = tid; i < HDIM * HDIM; i += 128) sW2[i] = W2[i];
    if (tid < HDIM) { sb1[tid] = b1[tid]; sg[tid] = g1[tid]; sb[tid] = bt1[tid]; sb2[tid] = b2[tid]; }
    __syncthreads();

    const size_t idx = (size_t)blockIdx.x * 128 + tid;  // token: b-fastest
    const int b  = (int)(idx & (BATCH - 1));
    const int tt = (int)(idx >> 9);

    float s[NSENS];
    const float* sp = sensors + ((size_t)b * TSTEPS + tt) * NSENS;
    #pragma unroll
    for (int k = 0; k < NSENS; ++k) s[k] = sp[k];

    float acc[HDIM];
    #pragma unroll
    for (int h = 0; h < HDIM; ++h) acc[h] = sb1[h];
    for (int k = 0; k < NSENS; ++k) {
        const float sv = s[k];
        #pragma unroll
        for (int h = 0; h < HDIM; ++h) acc[h] = fmaf(sv, sW1[k * HDIM + h], acc[h]);
    }
    // LayerNorm + GELU
    float m = 0.f;
    #pragma unroll
    for (int h = 0; h < HDIM; ++h) m += acc[h];
    m *= (1.0f / HDIM);
    float v = 0.f;
    #pragma unroll
    for (int h = 0; h < HDIM; ++h) { float d = acc[h] - m; v += d * d; }
    const float rstd = rsqrtf(v * (1.0f / HDIM) + 1e-5f);
    #pragma unroll
    for (int h = 0; h < HDIM; ++h) {
        float xh = (acc[h] - m) * rstd * sg[h] + sb[h];
        acc[h] = gelu_f(xh);
    }
    // second linear, emit bf16 pairs (coalesced: consecutive threads -> consecutive b)
    unsigned int* xo = (unsigned int*)(xin + ((size_t)tt * BATCH + b) * HDIM);
    for (int h2 = 0; h2 < HDIM; h2 += 2) {
        float o0 = sb2[h2], o1 = sb2[h2 + 1];
        #pragma unroll
        for (int h = 0; h < HDIM; ++h) {
            const float xh = acc[h];
            o0 = fmaf(xh, sW2[h * HDIM + h2], o0);
            o1 = fmaf(xh, sW2[h * HDIM + h2 + 1], o1);
        }
        xo[h2 >> 1] = (unsigned int)f2bf(o0) | ((unsigned int)f2bf(o1) << 16);
    }
}

// ---------------------------------------------------------------------------
// K3: persistent LSTM. grid = B/16 blocks, 256 threads (8 waves, 2/SIMD).
// Per layer-step: gates[16x256] = [x|h](16x128) @ WgT, via bf16 WMMA.
// wave w owns 2 N-tiles (cols 32w..32w+31); all 24 tile loads issued before
// the 8 WMMAs so one deep dscnt wait replaces 8 round-trip stalls.
// ---------------------------------------------------------------------------
__global__ void __launch_bounds__(256, 1) lstm_kernel(
        const ushort_t* __restrict__ xin,   // bf16 [T][B][H]
        const ushort_t* __restrict__ wt,    // bf16 [L][256][128]
        const float* __restrict__ bg,       // f32  [L][256]
        ushort_t* __restrict__ hout) {      // bf16 [B][T][H]
    extern __shared__ char smem[];
    ushort_t* wlds  = (ushort_t*)(smem);                       // [L][256][128]
    ushort_t* albuf = (ushort_t*)(smem + LDS_W);               // [L][16][128]: lo=input, hi=h
    float*    cbuf  = (float*)(smem + LDS_W + LDS_AL);         // [L][16][64]
    float*    gates = (float*)(smem + LDS_W + LDS_AL + LDS_C); // [16][256]
    float*    bgs   = (float*)(smem + LDS_W + LDS_AL + LDS_C + LDS_G); // [L][256]

    const int tid = threadIdx.x;
    const int b0  = blockIdx.x * 16;

    { // one-time init: weights + biases to LDS, h/c = 0
        const uint4* wsrc = (const uint4*)wt;
        uint4* wdst = (uint4*)wlds;
        for (int i = tid; i < LDS_W / 16; i += 256) wdst[i] = wsrc[i];
        const uint4 z = make_uint4(0u, 0u, 0u, 0u);
        uint4* az = (uint4*)albuf;
        for (int i = tid; i < LDS_AL / 16; i += 256) az[i] = z;
        uint4* cz = (uint4*)cbuf;
        for (int i = tid; i < LDS_C / 16; i += 256) cz[i] = z;
        const uint4* bsrc = (const uint4*)bg;
        uint4* bdst = (uint4*)bgs;
        for (int i = tid; i < LDS_BG / 16; i += 256) bdst[i] = bsrc[i];
    }
    __syncthreads();

    const int lane = tid & 31;
    const int wave = tid >> 5;           // 0..7, owns gate cols [32w, 32w+32)
    const int rA   = lane & 15;
    const int koff = (lane >> 4) << 3;   // lo lanes k+0, hi lanes k+8
    const int moff = (lane >> 4) << 3;   // C: hi lanes hold rows m+8
    const int er   = tid >> 4;           // elementwise row 0..15
    const int ej   = (tid & 15) << 2;    // elementwise col base 0..60, 4 elems

    // per-thread staging addresses for the async x_t tile copy (8B per lane)
    const unsigned xlds = (unsigned)(uintptr_t)(albuf + (size_t)er * 128 + ej);

    for (int t = 0; t < TSTEPS; ++t) {
        { // stage x_t tile (16 rows x 64 bf16 = 2KB) into layer-0 input half
          // via async global->LDS copy (ASYNCcnt path), then publish at barrier
            const ushort_t* gp = xin + ((size_t)t * BATCH + b0) * HDIM + tid * 4;
            asm volatile("global_load_async_to_lds_b64 %0, %1, off"
                         :: "v"(xlds), "v"(gp) : "memory");
            if (t + 1 < TSTEPS)
                __builtin_prefetch(xin + ((size_t)(t + 1) * BATCH + b0) * HDIM + tid * 4, 0, 0);
            asm volatile("s_wait_asynccnt 0x0" ::: "memory");
        }
        __syncthreads();

        for (int l = 0; l < NLAY; ++l) {
            const ushort_t* arow = albuf + l * (16 * 128) + rA * 128;
            const ushort_t* brow = wlds + l * (256 * 128) + (wave * 32 + rA) * 128;
            // issue all 24 b128 LDS loads up front, then compute
            v16bf A[4], B0[4], B1[4];
            #pragma unroll
            for (int kt = 0; kt < 4; ++kt) {
                const int kb = kt * 32 + koff;
                A[kt]  = ldAB(arow + kb);
                B0[kt] = ldAB(brow + kb);
                B1[kt] = ldAB(brow + 16 * 128 + kb);
            }
            v8f c0 = {}, c1 = {};
            #pragma unroll
            for (int kt = 0; kt < 4; ++kt) {
                c0 = __builtin_amdgcn_wmma_f32_16x16x32_bf16(false, A[kt], false, B0[kt], (short)0, c0, false, false);
                c1 = __builtin_amdgcn_wmma_f32_16x16x32_bf16(false, A[kt], false, B1[kt], (short)0, c1, false, false);
            }
            { // C layout: lane holds column n = lane&15, VGPR v holds row v(+8 hi lanes)
                const int nb = wave * 32 + (lane & 15);
                #pragma unroll
                for (int v = 0; v < 8; ++v) {
                    const int m = v + moff;
                    gates[m * 256 + nb]      = c0[v];
                    gates[m * 256 + nb + 16] = c1[v];
                }
            }
            __syncthreads();

            { // elementwise LSTM cell: 4 elements per thread
                const float* bgl = bgs + l * 256;
                float* cl = cbuf + l * (16 * 64) + er * 64 + ej;
                const float* gr = gates + er * 256;
                H4 hh;
                #pragma unroll
                for (int q = 0; q < 4; ++q) {
                    const int j = ej + q;
                    const float gi = gr[j]       + bgl[j];
                    const float gf = gr[64 + j]  + bgl[64 + j];
                    const float gg = gr[128 + j] + bgl[128 + j];
                    const float go = gr[192 + j] + bgl[192 + j];
                    const float cn = sigmoid_f(gf) * cl[q] + sigmoid_f(gi) * tanh_fast(gg);
                    cl[q] = cn;
                    hh.a[q] = f2bf(sigmoid_f(go) * tanh_fast(cn));
                }
                const uint2 hv = __builtin_bit_cast(uint2, hh);
                // persistent hidden state (K-half of this layer's A matrix)
                *(uint2*)(albuf + l * (16 * 128) + er * 128 + 64 + ej) = hv;
                if (l < NLAY - 1)   // next layer's input half
                    *(uint2*)(albuf + (l + 1) * (16 * 128) + er * 128 + ej) = hv;
                else                // final layer -> global bf16 [B][T][H]
                    *(uint2*)(hout + ((size_t)(b0 + er) * TSTEPS + t) * HDIM + ej) = hv;
            }
            __syncthreads();
        }
    }
}

// ---------------------------------------------------------------------------
// K4: y = GELU(LN(h3) @ W_out1 + b_out1) @ W_out2 + b_out2 -> fp32 [B][T][3]
// ---------------------------------------------------------------------------
__global__ void __launch_bounds__(128) out_proj_kernel(
        const ushort_t* __restrict__ h3,
        const float* __restrict__ g2, const float* __restrict__ bt2,
        const float* __restrict__ W1, const float* __restrict__ b1,
        const float* __restrict__ W2, const float* __restrict__ b2,
        float* __restrict__ y) {
    __shared__ float sW1[HDIM * HDIM];
    __shared__ float sW2[HDIM * ODIM];
    __shared__ float sg[HDIM], sb[HDIM], sb1[HDIM], sb2[ODIM];
    const int tid = threadIdx.x;
    for (int i = tid; i < HDIM * HDIM; i += 128) sW1[i] = W1[i];
    for (int i = tid; i < HDIM * ODIM; i += 128) sW2[i] = W2[i];
    if (tid < HDIM) { sg[tid] = g2[tid]; sb[tid] = bt2[tid]; sb1[tid] = b1[tid]; }
    if (tid < ODIM) sb2[tid] = b2[tid];
    __syncthreads();

    const size_t tok = (size_t)blockIdx.x * 128 + tid;  // tok = b*T + tt

    float xh[HDIM];
    { // 64 bf16 -> fp32
        const uint4* hp = (const uint4*)(h3 + tok * HDIM);
        #pragma unroll
        for (int i = 0; i < 8; ++i) {
            const uint4 q = hp[i];
            const unsigned int w[4] = { q.x, q.y, q.z, q.w };
            #pragma unroll
            for (int j = 0; j < 4; ++j) {
                xh[i * 8 + 2 * j]     = bf2f((unsigned short)(w[j] & 0xFFFFu));
                xh[i * 8 + 2 * j + 1] = bf2f((unsigned short)(w[j] >> 16));
            }
        }
    }
    float m = 0.f;
    #pragma unroll
    for (int h = 0; h < HDIM; ++h) m += xh[h];
    m *= (1.0f / HDIM);
    float v = 0.f;
    #pragma unroll
    for (int h = 0; h < HDIM; ++h) { float d = xh[h] - m; v += d * d; }
    const float rstd = rsqrtf(v * (1.0f / HDIM) + 1e-5f);
    #pragma unroll
    for (int h = 0; h < HDIM; ++h) xh[h] = (xh[h] - m) * rstd * sg[h] + sb[h];

    float y0 = sb2[0], y1 = sb2[1], y2 = sb2[2];
    for (int h2 = 0; h2 < HDIM; ++h2) {
        float o = sb1[h2];
        #pragma unroll
        for (int h = 0; h < HDIM; ++h) o = fmaf(xh[h], sW1[h * HDIM + h2], o);
        const float ge = gelu_f(o);
        y0 = fmaf(ge, sW2[h2 * ODIM + 0], y0);
        y1 = fmaf(ge, sW2[h2 * ODIM + 1], y1);
        y2 = fmaf(ge, sW2[h2 * ODIM + 2], y2);
    }
    float* yp = y + tok * ODIM;
    yp[0] = y0; yp[1] = y1; yp[2] = y2;
}

// ---------------------------------------------------------------------------
extern "C" void kernel_launch(void* const* d_in, const int* in_sizes, int n_in,
                              void* d_out, int out_size, void* d_ws, size_t ws_size,
                              hipStream_t stream) {
    (void)in_sizes; (void)n_in; (void)out_size; (void)ws_size;
    const float* sensors = (const float*)d_in[0];
    const float* W_in1   = (const float*)d_in[1];
    const float* b_in1   = (const float*)d_in[2];
    const float* ln1_g   = (const float*)d_in[3];
    const float* ln1_b   = (const float*)d_in[4];
    const float* W_in2   = (const float*)d_in[5];
    const float* b_in2   = (const float*)d_in[6];
    const float* Wg      = (const float*)d_in[7];
    const float* bg      = (const float*)d_in[8];
    const float* ln2_g   = (const float*)d_in[9];
    const float* ln2_b   = (const float*)d_in[10];
    const float* W_out1  = (const float*)d_in[11];
    const float* b_out1  = (const float*)d_in[12];
    const float* W_out2  = (const float*)d_in[13];
    const float* b_out2  = (const float*)d_in[14];
    float* y = (float*)d_out;

    char* ws = (char*)d_ws;
    ushort_t* xin = (ushort_t*)(ws + XIN_OFF);
    ushort_t* h3  = (ushort_t*)(ws + H3_OFF);
    ushort_t* wt  = (ushort_t*)(ws + WT_OFF);

    hipFuncSetAttribute((const void*)lstm_kernel,
                        hipFuncAttributeMaxDynamicSharedMemorySize, SMEM_BYTES);

    prep_weights_kernel<<<512, 256, 0, stream>>>(Wg, wt);
    input_proj_kernel<<<(BATCH * TSTEPS) / 128, 128, 0, stream>>>(
        sensors, W_in1, b_in1, ln1_g, ln1_b, W_in2, b_in2, xin);
    lstm_kernel<<<BATCH / 16, 256, SMEM_BYTES, stream>>>(xin, wt, bg, h3);
    out_proj_kernel<<<(BATCH * TSTEPS) / 128, 128, 0, stream>>>(
        h3, ln2_g, ln2_b, W_out1, b_out1, W_out2, b_out2, y);
}
